// FinalCompactModel_2997887172788
// MI455X (gfx1250) — compile-verified, matrix operand order checked by reading, not run on hardware
//
#include <hip/hip_runtime.h>
#include <math.h>

typedef unsigned short u16;
typedef __attribute__((ext_vector_type(8)))  u16   u16x8;
typedef __attribute__((ext_vector_type(16))) u16   u16x16;
typedef __attribute__((ext_vector_type(16))) __bf16 v16bf;
typedef __attribute__((ext_vector_type(8)))  float v8f;

constexpr int cB = 1024, cS = 100, cD = 96, cLOC = 64, cUE = 16, cTD = 16;
constexpr int cH = 4, cDH = 24, cFF = 192, cL = 40000;
constexpr int BS = cB * cS;

// ---------- helpers ----------
__device__ __forceinline__ u16 f2bf(float f) {
  unsigned u = __builtin_bit_cast(unsigned, f);
  u += 0x7FFFu + ((u >> 16) & 1u);          // round-to-nearest-even
  return (u16)(u >> 16);
}
__device__ __forceinline__ float gelu_exact(float x) {
  return 0.5f * x * (1.0f + erff(x * 0.70710678118f));
}

// ---------- fp32 -> bf16 convert ----------
__global__ void k_f2bf(const float* __restrict__ in, u16* __restrict__ out, int n) {
  int i = blockIdx.x * blockDim.x + threadIdx.x;
  if (i < n) out[i] = f2bf(in[i]);
}

// ---------- fused embed + temporal + input LN + positional encoding ----------
__global__ void k_front(const int* __restrict__ loc_seq, const int* __restrict__ user_seq,
                        const int* __restrict__ weekday, const float* __restrict__ start_min,
                        const float* __restrict__ dur, const float* __restrict__ diff,
                        const float* __restrict__ loc_table, const float* __restrict__ user_table,
                        const float* __restrict__ Wt, const float* __restrict__ bt,
                        const float* __restrict__ g_in, const float* __restrict__ b_in,
                        float* __restrict__ xf, u16* __restrict__ xb) {
  int idx = blockIdx.x * blockDim.x + threadIdx.x;   // b*S + s
  if (idx >= BS) return;
  int s = idx % cS;
  float v[cD];
  const float* lr = loc_table + (size_t)loc_seq[idx] * cLOC;
  #pragma unroll
  for (int i = 0; i < cLOC; ++i) v[i] = lr[i];
  const float* ur = user_table + (size_t)user_seq[idx] * cUE;
  #pragma unroll
  for (int i = 0; i < cUE; ++i) v[cLOC + i] = ur[i];
  // temporal features -> Linear(6 -> 16)
  float tr = start_min[idx] * (6.2831853072f / 1440.0f);   // (x/60)/24*2pi
  float wd = (float)weekday[idx] * (6.2831853072f / 7.0f);
  float f[6];
  f[0] = sinf(tr); f[1] = cosf(tr); f[2] = log1pf(dur[idx]) * 0.125f;
  f[3] = sinf(wd); f[4] = cosf(wd); f[5] = diff[idx] * (1.0f / 7.0f);
  #pragma unroll
  for (int i = 0; i < cTD; ++i) {
    float a = bt[i];
    #pragma unroll
    for (int j = 0; j < 6; ++j) a += Wt[i * 6 + j] * f[j];
    v[cLOC + cUE + i] = a;
  }
  // layernorm (population var) + PE
  float sum = 0.f, sq = 0.f;
  #pragma unroll
  for (int d = 0; d < cD; ++d) { sum += v[d]; sq += v[d] * v[d]; }
  float mean = sum * (1.0f / cD);
  float var  = sq * (1.0f / cD) - mean * mean;
  float inv  = rsqrtf(var + 1e-5f);
  float*  po = xf + (size_t)idx * cD;
  u16*    pb = xb + (size_t)idx * cD;
  #pragma unroll
  for (int d = 0; d < cD; ++d) {
    float freq = __expf(-0.0959410763f * (float)(d & ~1));   // log(1e4)/96
    float pe = (d & 1) ? cosf((float)s * freq) : sinf((float)s * freq);
    float y = (v[d] - mean) * inv * g_in[d] + b_in[d] + pe;
    po[d] = y;
    pb[d] = f2bf(y);
  }
}

// ---------- WMMA fragment loader (A and B share the K split per CDNA5 ISA 7.12.2) ----------
__device__ __forceinline__ v16bf load_frag(const u16* __restrict__ base, int ld, int lane) {
  // lanes 0-15: row (lane), K = {0..7, 16..23}; lanes 16-31: same rows, K = {8..15, 24..31}
  const u16* p = base + (size_t)(lane & 15) * ld + ((lane >> 4) << 3);
  u16x8 lo = *(const u16x8*)(p);
  u16x8 hi = *(const u16x8*)(p + 16);
  u16x16 w = __builtin_shufflevector(lo, hi, 0,1,2,3,4,5,6,7,8,9,10,11,12,13,14,15);
  return __builtin_bit_cast(v16bf, w);
}

// ---------- generic bf16 WMMA GEMM: C[M,N] = A[M,K] @ W[N,K]^T (+bias, opt GELU) ----------
__global__ void __launch_bounds__(256)
k_gemm_bf16(const u16* __restrict__ A, const u16* __restrict__ W,
            const float* __restrict__ bias, float* __restrict__ Cf, u16* __restrict__ Cb,
            int M, int N, int K, int gelu) {
  int lane = threadIdx.x & 31;
  int wave = blockIdx.x * (blockDim.x >> 5) + (threadIdx.x >> 5);
  int ntiles = N >> 4;
  int total  = (M >> 4) * ntiles;
  if (wave >= total) return;                  // wave-uniform: EXEC all-ones inside
  int m0 = (wave / ntiles) << 4;
  int n0 = (wave % ntiles) << 4;
  v8f acc = {};
  for (int k0 = 0; k0 < K; k0 += 32) {
    v16bf a = load_frag(A + (size_t)m0 * K + k0, K, lane);
    v16bf b = load_frag(W + (size_t)n0 * K + k0, K, lane);
    acc = __builtin_amdgcn_wmma_f32_16x16x32_bf16(false, a, false, b, (short)0, acc,
                                                  false, false);
  }
  int n  = n0 + (lane & 15);
  int mb = m0 + ((lane >> 4) << 3);
  float bv = bias ? bias[n] : 0.0f;
  #pragma unroll
  for (int r = 0; r < 8; ++r) {
    float y = acc[r] + bv;
    if (gelu) y = gelu_exact(y);
    size_t o = (size_t)(mb + r) * N + n;
    if (Cf) Cf[o] = y;
    if (Cb) Cb[o] = f2bf(y);
  }
}

// ---------- specialized predictor GEMM: M=1024, N=40000, K=192 ----------
// Each wave computes a 16x64 output strip: all 6 A fragments (48 VGPRs) are loaded once
// and reused across 4 consecutive N tiles -> 24 WMMAs per wave, 4x less A-frag traffic.
__global__ void __launch_bounds__(256)
k_gemm_pred(const u16* __restrict__ A, const u16* __restrict__ W,
            const float* __restrict__ bias, float* __restrict__ C) {
  constexpr int M = cB, N = cL, K = cFF;
  constexpr int KK = K / 32;                 // 6
  constexpr int nsuper = N / 64;             // 625 strips of 4 tiles
  constexpr int total  = (M / 16) * nsuper;  // 40000 waves
  int lane = threadIdx.x & 31;
  int wave = blockIdx.x * (blockDim.x >> 5) + (threadIdx.x >> 5);
  if (wave >= total) return;
  int m0 = (wave / nsuper) << 4;
  int n0 = (wave % nsuper) << 6;
  // warm L2 for the next strip of Wp2 this wave-group will touch (global_prefetch_b8)
  __builtin_prefetch(W + (size_t)(n0 + 64) * K, 0, 1);
  v16bf afrag[KK];
  #pragma unroll
  for (int kk = 0; kk < KK; ++kk)
    afrag[kk] = load_frag(A + (size_t)m0 * K + kk * 32, K, lane);
  #pragma unroll
  for (int t = 0; t < 4; ++t) {
    int nt0 = n0 + t * 16;
    v8f acc = {};
    #pragma unroll
    for (int kk = 0; kk < KK; ++kk) {
      v16bf b = load_frag(W + (size_t)nt0 * K + kk * 32, K, lane);
      acc = __builtin_amdgcn_wmma_f32_16x16x32_bf16(false, afrag[kk], false, b, (short)0,
                                                    acc, false, false);
    }
    int n  = nt0 + (lane & 15);
    int mb = m0 + ((lane >> 4) << 3);
    float bv = bias[n];
    #pragma unroll
    for (int r = 0; r < 8; ++r)
      C[(size_t)(mb + r) * N + n] = acc[r] + bv;
  }
}

// ---------- fp32 attention, one (batch, head) per block ----------
__global__ void k_attn(const float* __restrict__ qkv, const unsigned char* __restrict__ mask,
                       u16* __restrict__ ao) {
  int b = blockIdx.x / cH, h = blockIdx.x % cH;
  __shared__ float sk[cS * cDH];
  __shared__ float sv[cS * cDH];
  for (int i = threadIdx.x; i < cS * cDH; i += blockDim.x) {
    int s = i / cDH, j = i % cDH;
    size_t base = (size_t)(b * cS + s) * (3 * cD) + h * cDH + j;
    sk[i] = qkv[base + cD];
    sv[i] = qkv[base + 2 * cD];
  }
  __syncthreads();
  int t = threadIdx.x;
  if (t >= cS) return;
  float q[cDH];
  size_t qb = (size_t)(b * cS + t) * (3 * cD) + h * cDH;
  #pragma unroll
  for (int j = 0; j < cDH; ++j) q[j] = qkv[qb + j];
  const float scale = 0.20412414523f;         // 1/sqrt(24)
  float mx = -1e30f;
  for (int s2 = 0; s2 < cS; ++s2) {
    float d = 0.f;
    #pragma unroll
    for (int j = 0; j < cDH; ++j) d += q[j] * sk[s2 * cDH + j];
    d = mask[b * cS + s2] ? d * scale : -1e9f;
    mx = fmaxf(mx, d);
  }
  float acc[cDH];
  #pragma unroll
  for (int j = 0; j < cDH; ++j) acc[j] = 0.f;
  float sum = 0.f;
  for (int s2 = 0; s2 < cS; ++s2) {
    float d = 0.f;
    #pragma unroll
    for (int j = 0; j < cDH; ++j) d += q[j] * sk[s2 * cDH + j];
    d = mask[b * cS + s2] ? d * scale : -1e9f;
    float p = __expf(d - mx);
    sum += p;
    #pragma unroll
    for (int j = 0; j < cDH; ++j) acc[j] += p * sv[s2 * cDH + j];
  }
  float inv = 1.0f / sum;
  size_t ob = (size_t)(b * cS + t) * cD + h * cDH;
  #pragma unroll
  for (int j = 0; j < cDH; ++j) ao[ob + j] = f2bf(acc[j] * inv);
}

// ---------- residual add + layernorm ----------
__global__ void k_resln(const float* __restrict__ xa, const float* __restrict__ xr,
                        const float* __restrict__ g, const float* __restrict__ be,
                        float* __restrict__ of, u16* __restrict__ ob) {
  int idx = blockIdx.x * blockDim.x + threadIdx.x;
  if (idx >= BS) return;
  const float* pa = xa + (size_t)idx * cD;
  const float* pr = xr + (size_t)idx * cD;
  float v[cD]; float s = 0.f, sq = 0.f;
  #pragma unroll
  for (int d = 0; d < cD; ++d) { float t = pa[d] + pr[d]; v[d] = t; s += t; sq += t * t; }
  float mean = s * (1.0f / cD);
  float var  = sq * (1.0f / cD) - mean * mean;
  float inv  = rsqrtf(var + 1e-5f);
  #pragma unroll
  for (int d = 0; d < cD; ++d) {
    float y = (v[d] - mean) * inv * g[d] + be[d];
    of[(size_t)idx * cD + d] = y;
    if (ob) ob[(size_t)idx * cD + d] = f2bf(y);
  }
}

// ---------- gather last valid token, h = gelu(last @ Wp1^T + bp1) -> bf16 ----------
__global__ void k_pred_in(const float* __restrict__ x2, const unsigned char* __restrict__ mask,
                          const float* __restrict__ Wp1, const float* __restrict__ bp1,
                          u16* __restrict__ hb) {
  int b = blockIdx.x;
  int i = threadIdx.x;           // 0..191
  int len = 0;
  for (int s = 0; s < cS; ++s) len += mask[b * cS + s] ? 1 : 0;
  len -= 1;
  const float* xr = x2 + (size_t)(b * cS + len) * cD;
  float d = bp1[i];
  #pragma unroll
  for (int k = 0; k < cD; ++k) d += xr[k] * Wp1[i * cD + k];
  hb[(size_t)b * cFF + i] = f2bf(gelu_exact(d));
}

// ---------- decayed location histogram scatter ----------
__global__ void k_hist(const int* __restrict__ loc_seq, const unsigned char* __restrict__ mask,
                       const float* __restrict__ boost, const float* __restrict__ decay,
                       float* __restrict__ out) {
  int idx = blockIdx.x * blockDim.x + threadIdx.x;
  if (idx >= BS) return;
  if (!mask[idx]) return;
  int b = idx / cS, s = idx % cS;
  float w = boost[0] * powf(decay[0], (float)(cS - 1 - s));
  atomicAdd(out + (size_t)b * cL + loc_seq[idx], w);
}

// ---------- launcher ----------
extern "C" void kernel_launch(void* const* d_in, const int* in_sizes, int n_in,
                              void* d_out, int out_size, void* d_ws, size_t ws_size,
                              hipStream_t stream) {
  (void)in_sizes; (void)n_in; (void)out_size; (void)ws_size;
  const int*   loc_seq   = (const int*)d_in[0];
  const int*   user_seq  = (const int*)d_in[1];
  const int*   weekday   = (const int*)d_in[2];
  const float* start_min = (const float*)d_in[3];
  const float* dur       = (const float*)d_in[4];
  const float* diff      = (const float*)d_in[5];
  const unsigned char* mask = (const unsigned char*)d_in[6];
  const float* loc_table = (const float*)d_in[7];
  const float* user_table= (const float*)d_in[8];
  const float* Wt   = (const float*)d_in[9];
  const float* bt   = (const float*)d_in[10];
  const float* g_in = (const float*)d_in[11];
  const float* b_in = (const float*)d_in[12];
  const float* g1   = (const float*)d_in[13];
  const float* b1   = (const float*)d_in[14];
  const float* g2   = (const float*)d_in[15];
  const float* b2   = (const float*)d_in[16];
  const float* Wqkv = (const float*)d_in[17];
  const float* bqkv = (const float*)d_in[18];
  const float* Wo   = (const float*)d_in[19];
  const float* bo   = (const float*)d_in[20];
  const float* W1f  = (const float*)d_in[21];
  const float* b1f  = (const float*)d_in[22];
  const float* W2f  = (const float*)d_in[23];
  const float* b2f  = (const float*)d_in[24];
  const float* Wp1  = (const float*)d_in[25];
  const float* bp1  = (const float*)d_in[26];
  const float* Wp2  = (const float*)d_in[27];
  const float* bp2  = (const float*)d_in[28];
  const float* boost= (const float*)d_in[29];
  const float* decay= (const float*)d_in[30];
  float* out = (float*)d_out;

  // workspace carve-out (explicit reuse keeps footprint ~311 MB)
  size_t off = 0;
  auto take = [&](size_t bytes) -> char* {
    off = (off + 255) & ~(size_t)255;
    char* p = (char*)d_ws + off;
    off += bytes;
    return p;
  };
  float* xf    = (float*)take((size_t)BS * cD * 4);   // x after input LN+PE (reused for x2)
  u16*   xb    = (u16*)  take((size_t)BS * cD * 2);
  float* qkvf  = (float*)take((size_t)BS * 3 * cD * 4); // reused for FFN hidden (bf16)
  u16*   aob   = (u16*)  take((size_t)BS * cD * 2);
  float* aopf  = (float*)take((size_t)BS * cD * 4);   // reused for FFN2 output
  float* x1f   = (float*)take((size_t)BS * cD * 4);
  u16*   x1b   = (u16*)  take((size_t)BS * cD * 2);
  u16*   hb    = (u16*)  take((size_t)cB * cFF * 2);
  u16*   wqkvb = (u16*)  take((size_t)3 * cD * cD * 2);
  u16*   wob   = (u16*)  take((size_t)cD * cD * 2);
  u16*   w1fb  = (u16*)  take((size_t)cFF * cD * 2);
  u16*   w2fb  = (u16*)  take((size_t)cD * cFF * 2);
  u16*   wp2b  = (u16*)  take((size_t)cL * cFF * 2);
  u16*   ffb   = (u16*)qkvf;   // overlay: qkv dead after attention
  float* ff2f  = aopf;         // overlay: out-proj result dead after LN1
  float* x2f   = xf;           // overlay: pre-LN1 x dead after LN1

  auto cvt = [&](const float* src, u16* dst, int n) {
    k_f2bf<<<(n + 255) / 256, 256, 0, stream>>>(src, dst, n);
  };
  cvt(Wqkv, wqkvb, 3 * cD * cD);
  cvt(Wo,   wob,   cD * cD);
  cvt(W1f,  w1fb,  cFF * cD);
  cvt(W2f,  w2fb,  cD * cFF);
  cvt(Wp2,  wp2b,  cL * cFF);

  k_front<<<(BS + 255) / 256, 256, 0, stream>>>(loc_seq, user_seq, weekday, start_min, dur,
                                                diff, loc_table, user_table, Wt, bt, g_in,
                                                b_in, xf, xb);

  auto gemm = [&](const u16* A, const u16* W, const float* bias, float* Cf, u16* Cb,
                  int M, int N, int K, int gelu) {
    int tiles = (M / 16) * (N / 16);
    int blocks = (tiles + 7) / 8;             // 8 waves (256 thr) per block, tile per wave
    k_gemm_bf16<<<blocks, 256, 0, stream>>>(A, W, bias, Cf, Cb, M, N, K, gelu);
  };

  gemm(xb, wqkvb, bqkv, qkvf, nullptr, BS, 3 * cD, cD, 0);          // QKV projection
  k_attn<<<cB * cH, 128, 0, stream>>>(qkvf, mask, aob);             // attention (fp32)
  gemm(aob, wob, bo, aopf, nullptr, BS, cD, cD, 0);                 // output projection
  k_resln<<<(BS + 255) / 256, 256, 0, stream>>>(xf, aopf, g1, b1, x1f, x1b);
  gemm(x1b, w1fb, b1f, nullptr, ffb, BS, cFF, cD, 1);               // FFN1 + GELU -> bf16
  gemm(ffb, w2fb, b2f, ff2f, nullptr, BS, cD, cFF, 0);              // FFN2
  k_resln<<<(BS + 255) / 256, 256, 0, stream>>>(x1f, ff2f, g2, b2, x2f, nullptr);
  k_pred_in<<<cB, cFF, 0, stream>>>(x2f, mask, Wp1, bp1, hb);       // last token + GELU(Wp1)

  // predictor GEMM -> d_out (16x64 strip per wave, A frags register-resident)
  {
    constexpr int waves = (cB / 16) * (cL / 64);   // 40000
    k_gemm_pred<<<(waves + 7) / 8, 256, 0, stream>>>(hb, wp2b, bp2, out);
  }
  k_hist<<<(BS + 255) / 256, 256, 0, stream>>>(loc_seq, mask, boost, decay, out);
}
